// MSAAttention_24412594111333
// MI455X (gfx1250) — matrix operand order repaired and checked by reading, not compiled, not run on hardware
//
#include <hip/hip_runtime.h>
#include <cstdint>

typedef __attribute__((ext_vector_type(16))) _Float16 v16h;
typedef __attribute__((ext_vector_type(8)))  _Float16 v8h;
typedef __attribute__((ext_vector_type(8)))  float    v8f;
typedef __attribute__((ext_vector_type(4)))  float    f4;
typedef int v4i_vs __attribute__((vector_size(16)));   // matches builtin param type

constexpr int kS  = 128;   // N_seq
constexpr int kR  = 384;   // N_res
constexpr int kCM = 256;   // c_in
constexpr int kCZ = 128;   // pair channels
constexpr int kH  = 8;     // heads
constexpr int kP  = 1024;  // 4*H*D packed projection width
constexpr float kINF = 1e9f;
constexpr float kEPS = 1e-5f;

#if defined(__gfx1250__) && __has_builtin(__builtin_amdgcn_global_load_async_to_lds_b128)
#define HAVE_ASYNC_LDS 1
#else
#define HAVE_ASYNC_LDS 0
#endif

__device__ __forceinline__ v16h cat8(v8h a, v8h b) {
  return __builtin_shufflevector(a, b, 0,1,2,3,4,5,6,7,8,9,10,11,12,13,14,15);
}
__device__ __forceinline__ v8f vzero8() {
  v8f r = {0.f,0.f,0.f,0.f,0.f,0.f,0.f,0.f};
  return r;
}
__device__ __forceinline__ float sigmoidf_(float x) { return 1.f / (1.f + __expf(-x)); }

// 16-byte global -> LDS copy: async engine if available, else through VGPRs.
__device__ __forceinline__ void copy16_g2l(const void* g, void* l) {
#if HAVE_ASYNC_LDS
  __builtin_amdgcn_global_load_async_to_lds_b128(
      (__attribute__((address_space(1))) v4i_vs*)(void*)g,
      (__attribute__((address_space(3))) v4i_vs*)l, 0, 0);
#else
  *(v8h*)l = *(const v8h*)g;
#endif
}
__device__ __forceinline__ void async_wait0() {
#if HAVE_ASYNC_LDS
#if __has_builtin(__builtin_amdgcn_s_wait_asynccnt)
  __builtin_amdgcn_s_wait_asynccnt(0);
#else
  asm volatile("s_wait_asynccnt 0x0" ::: "memory");
#endif
#endif
}

// ---------------------------------------------------------------------------
// Kernel 1: pack weights to f16. Wqkvg[256][1024] (q scaled by 1/sqrt(32)),
// Wo16[256][256] (already [h*32+d][c] row-major in the source layout).
// ---------------------------------------------------------------------------
__global__ __launch_bounds__(256) void pack_weights(
    const float* __restrict__ wq, const float* __restrict__ wk,
    const float* __restrict__ wv, const float* __restrict__ wg,
    const float* __restrict__ wo,
    _Float16* __restrict__ Wqkvg, _Float16* __restrict__ Wo16) {
  int idx = blockIdx.x * 256 + threadIdx.x;
  const float qs = 0.17677669529663687f;  // 1/sqrt(32)
  if (idx < kCM * kP) {
    int c = idx / kP, col = idx % kP;
    int p = col >> 8, cd = col & 255;
    float v;
    if (p == 0)      v = wq[c * 256 + cd] * qs;
    else if (p == 1) v = wk[c * 256 + cd];
    else if (p == 2) v = wv[c * 256 + cd];
    else             v = wg[c * 256 + cd];
    Wqkvg[idx] = (_Float16)v;
  }
  if (idx < kCM * kCM) Wo16[idx] = (_Float16)wo[idx];
}

// ---------------------------------------------------------------------------
// Kernel 2: pair bias = einsum("ijc,ch->hij", LN(z), w_z).  One wave per (i,j).
// Memory-bound (75 MB of z); wave32 butterfly reductions over 128 channels.
// ---------------------------------------------------------------------------
__global__ __launch_bounds__(256) void zbias_kernel(
    const float* __restrict__ z, const float* __restrict__ lzw,
    const float* __restrict__ lzb, const float* __restrict__ wz,
    float* __restrict__ bias) {
  __shared__ float s_wz[kCZ * kH];
  __shared__ float s_w[kCZ], s_b[kCZ];
  int tid = threadIdx.x;
  for (int i = tid; i < kCZ * kH; i += 256) s_wz[i] = wz[i];
  if (tid < kCZ) { s_w[tid] = lzw[tid]; s_b[tid] = lzb[tid]; }
  __syncthreads();

  int flat = blockIdx.x * 8 + (tid >> 5);
  int lane = tid & 31;
  int i = flat / kR, j = flat % kR;

  f4 zv = *(const f4*)&z[((size_t)(i * kR + j)) * kCZ + lane * 4];
  float s1 = zv.x + zv.y + zv.z + zv.w;
  float s2 = zv.x*zv.x + zv.y*zv.y + zv.z*zv.z + zv.w*zv.w;
  for (int m = 16; m >= 1; m >>= 1) { s1 += __shfl_xor(s1, m); s2 += __shfl_xor(s2, m); }
  float mu = s1 * (1.f / 128.f);
  float var = s2 * (1.f / 128.f) - mu * mu;
  float rs = rsqrtf(var + kEPS);

  float zn[4];
  for (int k = 0; k < 4; k++)
    zn[k] = (zv[k] - mu) * rs * s_w[lane * 4 + k] + s_b[lane * 4 + k];

  for (int h = 0; h < kH; h++) {
    float p = 0.f;
    for (int k = 0; k < 4; k++) p += zn[k] * s_wz[(lane * 4 + k) * kH + h];
    for (int m = 16; m >= 1; m >>= 1) p += __shfl_xor(p, m);
    if (lane == h) bias[((size_t)(h * kR + i)) * kR + j] = p;
  }
}

// ---------------------------------------------------------------------------
// Kernel 3: fused LN(m) + QKVG projection GEMM.
// C[49152 x 1024] = LN(m)[49152 x 256] x Wqkvg[256 x 1024], f16 WMMA, f32 acc.
// Block: 256 threads (8 waves), tile 64(M) x 128(N); sigmoid on g columns.
// B tiles staged through the async-to-LDS engine.
// ---------------------------------------------------------------------------
__global__ __launch_bounds__(256) void qkvg_gemm(
    const float* __restrict__ m, const float* __restrict__ lmw,
    const float* __restrict__ lmb, const _Float16* __restrict__ W,
    const float* __restrict__ bg, _Float16* __restrict__ qkvg) {
  __shared__ _Float16 Asm[64][264];
  __shared__ _Float16 Bsm[32][136];
  __shared__ float s1[256], s2[256];
  __shared__ float smean[64], srstd[64];
  __shared__ float swl[256], sbl[256];

  int tid = threadIdx.x;
  int mBase = (blockIdx.x >> 3) * 64;
  int nBase = (blockIdx.x & 7) * 128;
  swl[tid] = lmw[tid]; sbl[tid] = lmb[tid];

  // LayerNorm statistics: 4 threads per row, 64 channels each.
  {
    int row = tid >> 2, part = tid & 3;
    const float* mp = &m[((size_t)(mBase + row)) * kCM + part * 64];
    float a = 0.f, b = 0.f;
    for (int i = 0; i < 64; i += 4) {
      f4 v = *(const f4*)&mp[i];
      a += v.x + v.y + v.z + v.w;
      b += v.x*v.x + v.y*v.y + v.z*v.z + v.w*v.w;
    }
    s1[tid] = a; s2[tid] = b;
  }
  __syncthreads();
  if (tid < 64) {
    float a = s1[tid*4] + s1[tid*4+1] + s1[tid*4+2] + s1[tid*4+3];
    float b = s2[tid*4] + s2[tid*4+1] + s2[tid*4+2] + s2[tid*4+3];
    float mu = a * (1.f / 256.f);
    float var = b * (1.f / 256.f) - mu * mu;
    smean[tid] = mu;
    srstd[tid] = rsqrtf(var + kEPS);
  }
  __syncthreads();
  // Normalize + convert A tile to f16 (coalesced: one row per iteration).
  for (int i = 0; i < 64; i++) {
    float v = m[((size_t)(mBase + i)) * kCM + tid];
    Asm[i][tid] = (_Float16)((v - smean[i]) * srstd[i] * swl[tid] + sbl[tid]);
  }

  int w = tid >> 5, lane = tid & 31;
  int lrow = lane & 15, hi = lane >> 4;
  int msub = w & 3, nhalf = w >> 2;
  v8f acc[4] = {vzero8(), vzero8(), vzero8(), vzero8()};

  for (int kc = 0; kc < 8; kc++) {
    __syncthreads();
    {
      int kr = tid >> 3, seg = tid & 7;
      const _Float16* wp = &W[(size_t)(kc * 32 + kr) * kP + nBase + seg * 16];
      copy16_g2l(wp,     &Bsm[kr][seg * 16]);
      copy16_g2l(wp + 8, &Bsm[kr][seg * 16 + 8]);
      if (kc < 7) __builtin_prefetch(wp + 32 * kP, 0, 0);  // next K-chunk
      async_wait0();
    }
    __syncthreads();
    v16h a = cat8(*(v8h*)&Asm[msub * 16 + lrow][kc * 32 + hi * 8],
                  *(v8h*)&Asm[msub * 16 + lrow][kc * 32 + 16 + hi * 8]);
    for (int t = 0; t < 4; t++) {
      v16h b;
      for (int e = 0; e < 8; e++) {
        b[e]     = Bsm[hi * 8 + e][(nhalf * 4 + t) * 16 + lrow];
        b[e + 8] = Bsm[16 + hi * 8 + e][(nhalf * 4 + t) * 16 + lrow];
      }
      acc[t] = __builtin_amdgcn_wmma_f32_16x16x32_f16(
          false, a, false, b, (short)0, acc[t], false, false);
    }
  }

  for (int t = 0; t < 4; t++) {
    int ncol = nBase + nhalf * 64 + t * 16 + lrow;
    for (int j = 0; j < 8; j++) {
      size_t grow = (size_t)(mBase + msub * 16 + j + 8 * hi);
      float v = acc[t][j];
      if (ncol >= 768) v = sigmoidf_(v + bg[ncol - 768]);  // gate pre-activation
      qkvg[grow * kP + ncol] = (_Float16)v;
    }
  }
}

// ---------------------------------------------------------------------------
// Kernel 4: attention.  Block = (s, h, 64-query chunk), 4 waves (wave32).
// Whole K^T (transposed on LDS fill) + V slice in LDS (V via async-to-LDS);
// logits in 98KB LDS scratch; softmax via cross-lane pair reduction;
// A.V through f16 WMMA.
// ---------------------------------------------------------------------------
__global__ __launch_bounds__(128) void attn_kernel(
    const _Float16* __restrict__ qkvg, const float* __restrict__ zbias,
    const float* __restrict__ mask, _Float16* __restrict__ og) {
  __shared__ _Float16 Kt[32][392];     // [d][r]
  __shared__ _Float16 Vs[384][40];     // [r][d]
  __shared__ float    Lg[4][16][392];  // per-wave logits [q][key]

  int tid = threadIdx.x;
  int b = blockIdx.x;
  int s = b / (kH * 6);
  int rem = b % (kH * 6);
  int h = rem / 6;
  int qBase = (rem % 6) * 64;

  const _Float16* kv = qkvg + (size_t)s * kR * kP;

  // V: async engine copies contiguous 16B chunks straight into LDS.
  for (int idx = tid; idx < kR * 4; idx += 128) {
    int r = idx >> 2, c = (idx & 3) * 8;
    copy16_g2l(kv + (size_t)r * kP + 512 + h * 32 + c, &Vs[r][c]);
  }
  // K: transposed on the fly (element-wise, must go through VGPRs).
  for (int idx = tid; idx < kR * 16; idx += 128) {
    int r = idx >> 4, dp = (idx & 15) * 2;
    const _Float16* kp = kv + (size_t)r * kP + 256 + h * 32 + dp;
    Kt[dp][r]     = kp[0];
    Kt[dp + 1][r] = kp[1];
  }
  async_wait0();
  __syncthreads();

  int w = tid >> 5, lane = tid & 31, lrow = lane & 15, hi = lane >> 4;
  int qTile = qBase + w * 16;

  // Q fragment: constant across the whole key loop (D=32 == WMMA K).
  const _Float16* qp = kv + (size_t)(qTile + lrow) * kP + h * 32 + hi * 8;
  v16h aq = cat8(*(const v8h*)qp, *(const v8h*)(qp + 16));

  const float* zb = zbias + ((size_t)h * kR) * kR;
  const float* mk = mask + (size_t)s * kR;

  // Phase 1: logits = QK^T + mask bias + pair bias -> LDS.
  for (int kt = 0; kt < 24; kt++) {
    v16h bk;
    for (int e = 0; e < 8; e++) {
      bk[e]     = Kt[hi * 8 + e][kt * 16 + lrow];
      bk[e + 8] = Kt[16 + hi * 8 + e][kt * 16 + lrow];
    }
    v8f c = __builtin_amdgcn_wmma_f32_16x16x32_f16(
        false, aq, false, bk, (short)0, vzero8(), false, false);
    int key = kt * 16 + lrow;
    float mbias = kINF * (mk[key] - 1.f);
    for (int j = 0; j < 8; j++) {
      int qrow = j + 8 * hi;
      Lg[w][qrow][key] = c[j] + mbias + zb[(size_t)(qTile + qrow) * kR + key];
    }
  }
  __syncthreads();

  // Phase 2: per-row max and exp-sum (lanes l and l^16 co-own row l&15).
  float lm = -3.0e38f;
  for (int i = 0; i < 192; i++) lm = fmaxf(lm, Lg[w][lrow][hi * 192 + i]);
  lm = fmaxf(lm, __shfl_xor(lm, 16));
  float ls = 0.f;
  for (int i = 0; i < 192; i++) ls += __expf(Lg[w][lrow][hi * 192 + i] - lm);
  ls += __shfl_xor(ls, 16);
  float ri = 1.f / ls;

  // Phase 3: o = softmax(logits) . V via WMMA; probs rebuilt on the fly.
  v8f o0 = vzero8(), o1 = vzero8();
  for (int c2 = 0; c2 < 12; c2++) {
    int base0 = c2 * 32 + hi * 8;
    int base1 = c2 * 32 + 16 + hi * 8;
    v16h ap;
    for (int e = 0; e < 8; e++) {
      ap[e]     = (_Float16)(__expf(Lg[w][lrow][base0 + e] - lm) * ri);
      ap[e + 8] = (_Float16)(__expf(Lg[w][lrow][base1 + e] - lm) * ri);
    }
    v16h b0, b1;
    for (int e = 0; e < 8; e++) {
      b0[e]     = Vs[base0 + e][lrow];
      b0[e + 8] = Vs[base1 + e][lrow];
      b1[e]     = Vs[base0 + e][16 + lrow];
      b1[e + 8] = Vs[base1 + e][16 + lrow];
    }
    o0 = __builtin_amdgcn_wmma_f32_16x16x32_f16(false, ap, false, b0, (short)0, o0, false, false);
    o1 = __builtin_amdgcn_wmma_f32_16x16x32_f16(false, ap, false, b1, (short)0, o1, false, false);
  }

  // Epilogue: gate with sigmoid(g) (precomputed in qkvg), store f16.
  for (int j = 0; j < 8; j++) {
    int q = qTile + j + 8 * hi;
    size_t rowoff = (size_t)(s * kR + q);
    int d0 = lrow, d1 = 16 + lrow;
    float g0 = (float)qkvg[rowoff * kP + 768 + h * 32 + d0];
    float g1 = (float)qkvg[rowoff * kP + 768 + h * 32 + d1];
    og[rowoff * kCM + h * 32 + d0] = (_Float16)(o0[j] * g0);
    og[rowoff * kCM + h * 32 + d1] = (_Float16)(o1[j] * g1);
  }
}

// ---------------------------------------------------------------------------
// Kernel 5: output projection GEMM.
// out[49152 x 256] = og[49152 x 256] x Wo[256 x 256] + b_o, f32 output.
// A and B tiles staged through the async-to-LDS engine.
// ---------------------------------------------------------------------------
__global__ __launch_bounds__(256) void out_gemm(
    const _Float16* __restrict__ og, const _Float16* __restrict__ Wo,
    const float* __restrict__ bo, float* __restrict__ out) {
  __shared__ _Float16 Asm[64][264];
  __shared__ _Float16 Bsm[32][136];

  int tid = threadIdx.x;
  int mBase = (blockIdx.x >> 1) * 64;
  int nBase = (blockIdx.x & 1) * 128;

  // A tile: 64 rows x 512B, 16B chunks via async engine.
  for (int i = 0; i < 8; i++) {
    int idx = tid + 256 * i;
    int row = idx >> 5, c = (idx & 31) * 8;
    copy16_g2l(&og[((size_t)(mBase + row)) * kCM + c], &Asm[row][c]);
  }
  async_wait0();

  int w = tid >> 5, lane = tid & 31, lrow = lane & 15, hi = lane >> 4;
  int msub = w & 3, nhalf = w >> 2;
  v8f acc[4] = {vzero8(), vzero8(), vzero8(), vzero8()};

  for (int kc = 0; kc < 8; kc++) {
    __syncthreads();
    {
      int kr = tid >> 3, seg = tid & 7;
      const _Float16* wp = &Wo[(size_t)(kc * 32 + kr) * kCM + nBase + seg * 16];
      copy16_g2l(wp,     &Bsm[kr][seg * 16]);
      copy16_g2l(wp + 8, &Bsm[kr][seg * 16 + 8]);
      async_wait0();
    }
    __syncthreads();
    v16h a = cat8(*(v8h*)&Asm[msub * 16 + lrow][kc * 32 + hi * 8],
                  *(v8h*)&Asm[msub * 16 + lrow][kc * 32 + 16 + hi * 8]);
    for (int t = 0; t < 4; t++) {
      v16h b;
      for (int e = 0; e < 8; e++) {
        b[e]     = Bsm[hi * 8 + e][(nhalf * 4 + t) * 16 + lrow];
        b[e + 8] = Bsm[16 + hi * 8 + e][(nhalf * 4 + t) * 16 + lrow];
      }
      acc[t] = __builtin_amdgcn_wmma_f32_16x16x32_f16(
          false, a, false, b, (short)0, acc[t], false, false);
    }
  }

  for (int t = 0; t < 4; t++) {
    int ncol = nBase + nhalf * 64 + t * 16 + lrow;
    float bb = bo[ncol];
    for (int j = 0; j < 8; j++) {
      size_t grow = (size_t)(mBase + msub * 16 + j + 8 * hi);
      out[grow * kCM + ncol] = acc[t][j] + bb;
    }
  }
}

// ---------------------------------------------------------------------------
extern "C" void kernel_launch(void* const* d_in, const int* in_sizes, int n_in,
                              void* d_out, int out_size, void* d_ws, size_t ws_size,
                              hipStream_t stream) {
  const float* m      = (const float*)d_in[0];
  const float* z      = (const float*)d_in[1];
  const float* mask   = (const float*)d_in[2];
  const float* ln_m_w = (const float*)d_in[3];
  const float* ln_m_b = (const float*)d_in[4];
  const float* ln_z_w = (const float*)d_in[5];
  const float* ln_z_b = (const float*)d_in[6];
  const float* w_z    = (const float*)d_in[7];
  const float* w_q    = (const float*)d_in[8];
  const float* w_k    = (const float*)d_in[9];
  const float* w_v    = (const float*)d_in[10];
  const float* w_g    = (const float*)d_in[11];
  const float* b_g    = (const float*)d_in[12];
  const float* w_o    = (const float*)d_in[13];
  const float* b_o    = (const float*)d_in[14];
  float* out = (float*)d_out;

  char* ws = (char*)d_ws;
  size_t off = 0;
  auto carve = [&](size_t bytes) -> void* {
    void* p = ws + off;
    off += (bytes + 255) & ~(size_t)255;
    return p;
  };
  float*    zb    = (float*)   carve((size_t)kH * kR * kR * sizeof(float));      // 4.7 MB
  _Float16* Wqkvg = (_Float16*)carve((size_t)kCM * kP * sizeof(_Float16));       // 0.5 MB
  _Float16* Wo16  = (_Float16*)carve((size_t)kCM * kCM * sizeof(_Float16));      // 0.13 MB
  _Float16* qkvg  = (_Float16*)carve((size_t)kS * kR * kP * sizeof(_Float16));   // 100 MB
  _Float16* og    = (_Float16*)carve((size_t)kS * kR * kCM * sizeof(_Float16));  // 25 MB

  pack_weights<<<dim3(1024), dim3(256), 0, stream>>>(w_q, w_k, w_v, w_g, w_o, Wqkvg, Wo16);
  zbias_kernel<<<dim3(kR * kR / 8), dim3(256), 0, stream>>>(z, ln_z_w, ln_z_b, w_z, zb);
  qkvg_gemm<<<dim3((kS * kR / 64) * 8), dim3(256), 0, stream>>>(m, ln_m_w, ln_m_b, Wqkvg, b_g, qkvg);
  attn_kernel<<<dim3(kS * kH * 6), dim3(128), 0, stream>>>(qkvg, zb, mask, og);
  out_gemm<<<dim3((kS * kR / 64) * 2), dim3(256), 0, stream>>>(og, Wo16, b_o, out);
}
